// MoEknownCov_11192684774076
// MI455X (gfx1250) — compile-verified, matrix operand order checked by reading, not run on hardware
//
#include <hip/hip_runtime.h>
#include <stdint.h>
#include <stddef.h>

// ---------------- types (plain ext-vectors; safe inside unions) -----------
typedef __attribute__((ext_vector_type(16))) __bf16   v16bf;
typedef __attribute__((ext_vector_type(8)))  float    v8f;
typedef __attribute__((ext_vector_type(4))) unsigned u32x4;
typedef __attribute__((ext_vector_type(2))) unsigned u32x2;
typedef __attribute__((ext_vector_type(4)))  float    f32x4;

union FragU {           // one WMMA 16x32 bf16 operand fragment (8 VGPRs/lane)
    v16bf v;
    u32x4 u[2];
};

#define E_    16
#define DIN_  128
#define H_    512
#define DOUT_ 64
#define BT    128      // tokens per block

// fp32 -> bf16 round-to-nearest-even
__device__ __forceinline__ unsigned f2bf(float f) {
    unsigned u = __float_as_uint(f);
    u = u + 0x7FFFu + ((u >> 16) & 1u);
    return u >> 16;
}
__device__ __forceinline__ unsigned pk2(float a, float b) {
    return f2bf(a) | (f2bf(b) << 16);
}

// -------------------- pre-pass: W [E][K][N] fp32 -> Wt [E][N][K] bf16 -----
__global__ void cvt_w_kernel(const float* __restrict__ W,
                             unsigned short* __restrict__ Wt,
                             int K, int N, long total /* = E*N*K/2 */) {
    long idx = (long)blockIdx.x * blockDim.x + threadIdx.x;
    if (idx >= total) return;
    int  kh = K >> 1;
    long kk = (idx % kh) * 2;
    long t  = idx / kh;
    int  n  = (int)(t % N);
    int  e  = (int)(t / N);
    const float* src = W + ((size_t)e * K + kk) * N + n;
    unsigned p = pk2(src[0], src[N]);           // k and k+1 for column n
    *(unsigned*)&Wt[((size_t)e * N + n) * K + kk] = p;
}

__global__ void zero_out_kernel(float* __restrict__ o, int n4) {
    int i = blockIdx.x * blockDim.x + threadIdx.x;
    if (i < n4) ((f32x4*)o)[i] = f32x4{0.f, 0.f, 0.f, 0.f};
}

// -------- fused layer: hout[t][f] = act(Wt[f][:]·hin[t][:] + bias[f]) -----
// Register-blocked NFT feature-tiles x NTT token-tiles; bias rides in C-init.
template<int KD, int NFT, int NTT, bool RELU>
__device__ __forceinline__ void mlp_layer(const unsigned short* __restrict__ Wt,
                                          const float* __restrict__ bias,
                                          const unsigned short* hin,
                                          unsigned short* hout, int ldo,
                                          int ftile0, int lane) {
    const int hi = lane >> 4;        // half-wave selects K sub-chunk
    const int tl = lane & 15;        // token (B operand N index) / A row (M)
    v8f biasv[NFT];
#pragma unroll
    for (int i = 0; i < NFT; ++i)
        biasv[i] = *(const v8f*)(bias + (ftile0 + i) * 16 + hi * 8);

    for (int tg = 0; tg < BT / 16; tg += NTT) {
        v8f acc[NFT][NTT];
#pragma unroll
        for (int i = 0; i < NFT; ++i)
#pragma unroll
            for (int j = 0; j < NTT; ++j) acc[i][j] = biasv[i];

        for (int k = 0; k < KD; k += 32) {
            FragU b[NTT];
#pragma unroll
            for (int j = 0; j < NTT; ++j) {
                const unsigned short* hrow =
                    hin + (size_t)((tg + j) * 16 + tl) * KD + hi * 8 + k;
                b[j].u[0] = *(const u32x4*)(hrow);
                b[j].u[1] = *(const u32x4*)(hrow + 16);
            }
#pragma unroll
            for (int i = 0; i < NFT; ++i) {
                const unsigned short* arow =
                    Wt + (size_t)((ftile0 + i) * 16 + tl) * KD + hi * 8 + k;
                FragU a;
                a.u[0] = *(const u32x4*)(arow);
                a.u[1] = *(const u32x4*)(arow + 16);
#pragma unroll
                for (int j = 0; j < NTT; ++j)
                    acc[i][j] = __builtin_amdgcn_wmma_f32_16x16x32_bf16(
                        false, a.v, false, b[j].v, (short)0, acc[i][j],
                        false, false);
            }
        }
#pragma unroll
        for (int i = 0; i < NFT; ++i) {
            const int f0 = (ftile0 + i) * 16 + hi * 8;
#pragma unroll
            for (int j = 0; j < NTT; ++j) {
                const int t = (tg + j) * 16 + tl;
                float vv[8];
#pragma unroll
                for (int r = 0; r < 8; ++r)
                    vv[r] = RELU ? fmaxf(acc[i][j][r], 0.f) : acc[i][j][r];
                u32x4 p;
                p.x = pk2(vv[0], vv[1]); p.y = pk2(vv[2], vv[3]);
                p.z = pk2(vv[4], vv[5]); p.w = pk2(vv[6], vv[7]);
                *(u32x4*)(hout + (size_t)t * ldo + f0) = p;  // ds_store_b128
            }
        }
    }
}

// -------------------- main fused MoE MLP kernel ---------------------------
__global__ __launch_bounds__(256, 1)
void moe_mlp_kernel(const float* __restrict__ x,
                    const unsigned short* __restrict__ W1t, const float* __restrict__ b1,
                    const unsigned short* __restrict__ W2t, const float* __restrict__ b2,
                    const unsigned short* __restrict__ W3t, const float* __restrict__ b3,
                    float* __restrict__ out) {
    __shared__ unsigned short h1[BT * H_];   // 128 KB
    __shared__ unsigned short h2[BT * H_];   // 128 KB (first 32 KB doubles as x stage)
    unsigned short* xs = h2;                 // xs dead before layer-2 writes h2

    const int tb   = blockIdx.x & 63;        // token tile   (consecutive blocks:
    const int e    = blockIdx.x >> 6;        //  same expert -> atomics uncontended)
    const int tid  = threadIdx.x;
    const int lane = tid & 31;
    const int w    = tid >> 5;               // wave id 0..7

    // prefetch this expert's 512 KB W2 slice toward L2
    const unsigned short* w2e = W2t + (size_t)e * H_ * H_;
#pragma unroll
    for (int i = 0; i < 8; ++i)
        __builtin_prefetch(w2e + ((size_t)(i * 256 + tid)) * 128, 0, 0);

    // stage x tile [BT][DIN] fp32 -> bf16 LDS
    const float* xb = x + (size_t)tb * BT * E_ * DIN_ + (size_t)e * DIN_;
#pragma unroll
    for (int i = 0; i < 16; ++i) {
        int g  = i * 256 + tid;          // 4096 float4 groups
        int t  = g >> 5;
        int kg = (g & 31) << 2;
        f32x4 vv = *(const f32x4*)(xb + (size_t)t * (E_ * DIN_) + kg);
        *(u32x2*)&xs[t * DIN_ + kg] = u32x2{pk2(vv.x, vv.y), pk2(vv.z, vv.w)};
    }
    __syncthreads();

    // layer 1: h1[t][f] = relu(W1t[f][:]·x[t][:] + b1)     (K=128, NF=512)
    // two SEQUENTIAL f-groups (unroll 1!) keep live VGPRs < 256
#pragma unroll 1
    for (int fg = 0; fg < 2; ++fg)
        mlp_layer<DIN_, 2, 4, true>(W1t + (size_t)e * H_ * DIN_, b1 + e * H_,
                                    xs, h1, H_, w * 4 + fg * 2, lane);
    __syncthreads();

    // layer 2: h2[t][f] = relu(W2t[f][:]·h1[t][:] + b2)    (K=512, NF=512)
#pragma unroll 1
    for (int fg = 0; fg < 2; ++fg)
        mlp_layer<H_, 2, 4, true>(w2e, b2 + e * H_, h1, h2, H_,
                                  w * 4 + fg * 2, lane);
    __syncthreads();

    // layer 3: out[t][f] += W3t[f][:]·h2[t][:] + b3        (K=512, NF=64)
    {
        const int hi = lane >> 4, tl = lane & 15;
        const int ft = w & 3, tt0 = (w >> 2) * 4;
        const int f0 = ft * 16 + hi * 8;
        v8f bv = *(const v8f*)(b3 + e * DOUT_ + f0);
        v8f acc[4];
#pragma unroll
        for (int j = 0; j < 4; ++j) acc[j] = bv;
        const unsigned short* arowb =
            W3t + ((size_t)e * DOUT_ + ft * 16 + tl) * H_ + hi * 8;
        for (int k = 0; k < H_; k += 32) {
            FragU a;
            a.u[0] = *(const u32x4*)(arowb + k);
            a.u[1] = *(const u32x4*)(arowb + k + 16);
#pragma unroll
            for (int j = 0; j < 4; ++j) {
                const unsigned short* hrow =
                    h2 + (size_t)((tt0 + j) * 16 + tl) * H_ + hi * 8 + k;
                FragU b;
                b.u[0] = *(const u32x4*)(hrow);
                b.u[1] = *(const u32x4*)(hrow + 16);
                acc[j] = __builtin_amdgcn_wmma_f32_16x16x32_bf16(
                    false, a.v, false, b.v, (short)0, acc[j], false, false);
            }
        }
#pragma unroll
        for (int j = 0; j < 4; ++j) {
            const int t = (tt0 + j) * 16 + tl;
            float* orow = out + (size_t)(tb * BT + t) * DOUT_ + f0;
#pragma unroll
            for (int r = 0; r < 8; ++r)
                atomicAdd(orow + r, acc[j][r]);   // sum over experts
        }
    }
}

// -------------------- host launcher ---------------------------------------
extern "C" void kernel_launch(void* const* d_in, const int* in_sizes, int n_in,
                              void* d_out, int out_size, void* d_ws, size_t ws_size,
                              hipStream_t stream) {
    const float* x  = (const float*)d_in[0];
    const float* W1 = (const float*)d_in[1];
    const float* b1 = (const float*)d_in[2];
    const float* W2 = (const float*)d_in[3];
    const float* b2 = (const float*)d_in[4];
    const float* W3 = (const float*)d_in[5];
    const float* b3 = (const float*)d_in[6];
    float* out = (float*)d_out;

    // workspace layout: bf16 transposed weights [E][N][K]
    unsigned short* W1t = (unsigned short*)d_ws;                       // 16*512*128
    unsigned short* W2t = W1t + (size_t)E_ * H_ * DIN_;                // 16*512*512
    unsigned short* W3t = W2t + (size_t)E_ * H_ * H_;                  // 16*64*512

    {
        long t1 = (long)E_ * H_ * DIN_ / 2;
        cvt_w_kernel<<<(unsigned)((t1 + 255) / 256), 256, 0, stream>>>(W1, W1t, DIN_, H_, t1);
        long t2 = (long)E_ * H_ * H_ / 2;
        cvt_w_kernel<<<(unsigned)((t2 + 255) / 256), 256, 0, stream>>>(W2, W2t, H_, H_, t2);
        long t3 = (long)E_ * DOUT_ * H_ / 2;
        cvt_w_kernel<<<(unsigned)((t3 + 255) / 256), 256, 0, stream>>>(W3, W3t, H_, DOUT_, t3);
    }

    // zero output (expert-sum accumulator), then fused MLP
    int n4 = (8192 * DOUT_) / 4;
    zero_out_kernel<<<(n4 + 255) / 256, 256, 0, stream>>>(out, n4);

    dim3 grid((8192 / BT) * E_);   // 64 token tiles * 16 experts = 1024 blocks
    moe_mlp_kernel<<<grid, 256, 0, stream>>>(x, W1t, b1, W2t, b2, W3t, b3, out);
}
